// Related_55688545960493
// MI455X (gfx1250) — matrix operand order repaired and checked by reading, not compiled, not run on hardware
//
#include <hip/hip_runtime.h>
#include <hip/hip_bf16.h>

typedef __attribute__((ext_vector_type(16))) _Float16 v16h;
typedef __attribute__((ext_vector_type(8)))  _Float16 v8h;
typedef __attribute__((ext_vector_type(8)))  float    v8f;

#define BB   32
#define LL   8192
#define FF   64
#define NP   (BB * FF)          // 2048 independent (batch,feature) pairs

// LDS staging: zero-padded fp16 x and reversed y.
// A reads xp[mb + k' - 16i]          -> pad front 240, back 288
// B reads yp[mb + s0{a,b} + k' + j] over the union k-range of a tile PAIR
//   (s0b = s0a + 256): index in [-302, L+526] -> pad front 304, back 560
#define XPAD  240
#define XLEN  (XPAD + LL + 288)   // 8720 halfs
#define YPAD  304
#define YLEN  (YPAD + LL + 560)   // 9056 halfs

// A fragment: lane holds A[iRow, kbA..kbA+15]
// -> two 16B-aligned LDS vector loads, 16 bytes apart.
static __device__ __forceinline__ v16h load_a_frag(const _Float16* s_x, int idx) {
    v8h lo = *(const v8h*)(s_x + idx);
    v8h hi = *(const v8h*)(s_x + idx + 8);
    return __builtin_shufflevector(lo, hi,
        0, 1, 2, 3, 4, 5, 6, 7, 8, 9, 10, 11, 12, 13, 14, 15);
}

// B fragment: lane holds 16 consecutive halfs (compiler emits 2x ds_load_b128).
static __device__ __forceinline__ v16h load_b_frag(const _Float16* s_y, int idx) {
    v16h b;
    #pragma unroll
    for (int q = 0; q < 16; ++q) b[q] = s_y[idx + q];
    return b;
}

#define WMMA(a, b, c) __builtin_amdgcn_wmma_f32_16x16x32_f16( \
        false, (a), false, (b), (short)0, (c), false, false)

__global__ __launch_bounds__(256) void conv_max_kernel(
    const float* __restrict__ in1, const float* __restrict__ in2,
    float* __restrict__ pairSum, float* __restrict__ pairMax)
{
    __shared__ __attribute__((aligned(16))) _Float16 s_x[XLEN];
    __shared__ __attribute__((aligned(16))) _Float16 s_y[YLEN];
    __shared__ float s_red[2];
    __shared__ float s_wmax[8];

    const int p   = blockIdx.x;          // pair index = b*F + f
    const int b   = p >> 6;
    const int f   = p & (FF - 1);
    const int tid = threadIdx.x;

    const float* xg = in1 + (size_t)b * LL * FF + f;
    const float* yg = in2 + (size_t)b * LL * FF + f;

    // ---- stage: zero pads, load + convert fp32->fp16, reverse y, partial sums
    for (int t = tid; t < XLEN; t += 256) s_x[t] = (_Float16)0.0f;
    for (int t = tid; t < YLEN; t += 256) s_y[t] = (_Float16)0.0f;
    if (tid < 2) s_red[tid] = 0.0f;
    __syncthreads();

    float sx = 0.0f, sy = 0.0f;
    for (int l = tid; l < LL; l += 256) {
        float xv = xg[(size_t)l * FF];
        float yv = yg[(size_t)l * FF];
        sx += xv; sy += yv;
        s_x[XPAD + l]            = (_Float16)xv;
        s_y[YPAD + (LL - 1 - l)] = (_Float16)yv;   // reversed y
    }
    atomicAdd(&s_red[0], sx);
    atomicAdd(&s_red[1], sy);
    __syncthreads();

    // ---- WMMA sliding-correlation: C[i][j] = z[s0 + 16*i + j]
    const int wave = tid >> 5;
    const int lane = tid & 31;
    const int iRow = lane & 15;                 // A: matrix row
    const int kbA  = (lane < 16) ? 0 : 16;      // A: per-half-wave K base
    const int jCol = lane & 15;                 // B: matrix column
    const int kbB  = (lane < 16) ? 0 : 16;      // B: per-half-wave K base

    const int axBase = XPAD - 16 * iRow + kbA;  // tile-independent! (16B aligned)
    float vmax = -3.4e38f;

    // 32 tile-pairs; wave w takes pair groups g = w, w+8, w+16, w+24
    // (uniform sampling of the triangular trip-count profile -> balanced).
    for (int k = 0; k < 4; ++k) {
        const int g   = wave + 8 * k;
        const int s0a = -8192 + 512 * g;       // tile 2g
        const int s0b = s0a + 256;             // tile 2g+1

        // union of both tiles' trimmed k-ranges (wave-uniform -> scalar loop)
        int lo = (s0b < -15) ? ((-s0b - 15) & ~31) : 0;
        int hi = LL;
        if (LL - s0a < hi) hi = LL - s0a;      // min(L1, L2 - s0a)
        int mb_hi = (hi + 241 + 31) & ~31;     // exclusive, mult of 32
        lo    = __builtin_amdgcn_readfirstlane(lo);
        mb_hi = __builtin_amdgcn_readfirstlane(mb_hi);

        const int byA = YPAD + s0a + kbB + jCol;
        const int byB = YPAD + s0b + kbB + jCol;

        v8f c0 = {}, c1 = {};

        // unroll-2 ping-pong: loads write directly into the buffer consumed
        // two WMMAs later -> no phi copies, partial dscnt waits.
        int mb = lo;
        v16h A0  = load_a_frag(s_x, axBase + mb);
        v16h B0a = load_b_frag(s_y, byA + mb);
        v16h B0b = load_b_frag(s_y, byB + mb);
        for (; mb + 64 < mb_hi; mb += 64) {
            v16h A1  = load_a_frag(s_x, axBase + mb + 32);
            v16h B1a = load_b_frag(s_y, byA + mb + 32);
            v16h B1b = load_b_frag(s_y, byB + mb + 32);
            c0 = WMMA(A0, B0a, c0);
            c1 = WMMA(A0, B0b, c1);
            A0  = load_a_frag(s_x, axBase + mb + 64);
            B0a = load_b_frag(s_y, byA + mb + 64);
            B0b = load_b_frag(s_y, byB + mb + 64);
            c0 = WMMA(A1, B1a, c0);
            c1 = WMMA(A1, B1b, c1);
        }
        if (mb + 32 < mb_hi) {                 // 2 steps left
            v16h A1  = load_a_frag(s_x, axBase + mb + 32);
            v16h B1a = load_b_frag(s_y, byA + mb + 32);
            v16h B1b = load_b_frag(s_y, byB + mb + 32);
            c0 = WMMA(A0, B0a, c0);
            c1 = WMMA(A0, B0b, c1);
            c0 = WMMA(A1, B1a, c0);
            c1 = WMMA(A1, B1b, c1);
        } else {                               // 1 step left
            c0 = WMMA(A0, B0a, c0);
            c1 = WMMA(A0, B0b, c1);
        }

        #pragma unroll
        for (int r = 0; r < 8; ++r) {
            vmax = fmaxf(vmax, c0[r]);
            vmax = fmaxf(vmax, c1[r]);
        }
    }

    // ---- reduce max: wave32 shuffle tree, then across the 8 waves
    for (int off = 16; off > 0; off >>= 1)
        vmax = fmaxf(vmax, __shfl_xor(vmax, off, 32));
    if (lane == 0) s_wmax[wave] = vmax;
    __syncthreads();

    if (tid == 0) {
        float m = s_wmax[0];
        #pragma unroll
        for (int w = 1; w < 8; ++w) m = fmaxf(m, s_wmax[w]);
        pairMax[p] = m;
        pairSum[p] = s_red[0] * s_red[1];   // sum(conv) = sum(x)*sum(y)
    }
}

__global__ __launch_bounds__(256) void finalize_kernel(
    const float* __restrict__ pairSum, const float* __restrict__ pairMax,
    float* __restrict__ out)
{
    __shared__ float red[256];
    const int tid = threadIdx.x;
    float acc = 0.0f;
    for (int p = tid; p < NP; p += 256) acc += pairSum[p] / pairMax[p];
    red[tid] = acc;
    __syncthreads();
    for (int s = 128; s > 0; s >>= 1) {
        if (tid < s) red[tid] += red[tid + s];
        __syncthreads();
    }
    if (tid == 0) out[0] = red[0];
}

extern "C" void kernel_launch(void* const* d_in, const int* in_sizes, int n_in,
                              void* d_out, int out_size, void* d_ws, size_t ws_size,
                              hipStream_t stream)
{
    (void)in_sizes; (void)n_in; (void)out_size; (void)ws_size;
    const float* a1 = (const float*)d_in[0];
    const float* a2 = (const float*)d_in[1];
    float* pairSum = (float*)d_ws;          // NP floats
    float* pairMax = pairSum + NP;          // NP floats

    conv_max_kernel<<<NP, 256, 0, stream>>>(a1, a2, pairSum, pairMax);
    finalize_kernel<<<1, 256, 0, stream>>>(pairSum, pairMax, (float*)d_out);
}
